// ProcessGCN_43722767073848
// MI455X (gfx1250) — matrix operand order repaired and checked by reading, not compiled, not run on hardware
//
#include <hip/hip_runtime.h>
#include <hip/hip_bf16.h>
#include <stdint.h>

#define NN 50000
#define NE 400000

typedef __attribute__((ext_vector_type(16))) __bf16 v16bf;
typedef __attribute__((ext_vector_type(8)))  float  v8f;

union ABu { v16bf v; uint32_t w[8]; uint4 q[2]; };

__device__ __forceinline__ unsigned short f2bf(float f) {
  union { float f; uint32_t u; } x; x.f = f;
  uint32_t u = x.u;
  u += 0x7FFFu + ((u >> 16) & 1u);          // round to nearest even
  return (unsigned short)(u >> 16);
}

// ---------------- small utility kernels ----------------

__global__ void f32_to_bf16_kernel(const float* __restrict__ in,
                                   unsigned short* __restrict__ out, long long n) {
  long long i = (long long)blockIdx.x * blockDim.x + threadIdx.x;
  if (i < n) out[i] = f2bf(in[i]);
}

__global__ void deg_init_kernel(float* __restrict__ deg, int n) {
  int i = blockIdx.x * blockDim.x + threadIdx.x;
  if (i < n) deg[i] = 1.0f;                  // self-loop contributes 1 to every node
}

__global__ void deg_edges_kernel(const int* __restrict__ ei, float* __restrict__ deg, int E) {
  int e = blockIdx.x * blockDim.x + threadIdx.x;
  if (e < E) atomicAdd(&deg[ei[E + e]], 1.0f);   // dst row of edge_index
}

__global__ void dinv_finish_kernel(float* __restrict__ deg, int n) {
  int i = blockIdx.x * blockDim.x + threadIdx.x;
  if (i < n) deg[i] = rsqrtf(deg[i]);        // deg >= 1 always (self loops)
}

// Pack W[CIN,COUT] (f32, row-major) into per-(ktile,ntile) WMMA B-fragments (bf16):
// Wpk[((kt*NT + nt)*32 + lane)*8 + w] packs K = kt*32 + 16*(lane>=16) + 2w (+1), n = nt*16 + (lane&15)
__global__ void pack_w_kernel(const float* __restrict__ W, uint32_t* __restrict__ Wpk,
                              int CIN, int COUT, int NT, int total) {
  int idx = blockIdx.x * blockDim.x + threadIdx.x;
  if (idx >= total) return;
  int w    = idx & 7;
  int lane = (idx >> 3) & 31;
  int tile = idx >> 8;
  int nt   = tile % NT;
  int kt   = tile / NT;
  int n  = nt * 16 + (lane & 15);
  int kb = kt * 32 + ((lane >> 4) << 4) + (w << 1);
  uint32_t lo = (kb     < CIN && n < COUT) ? (uint32_t)f2bf(W[(size_t)kb       * COUT + n]) : 0u;
  uint32_t hi = (kb + 1 < CIN && n < COUT) ? (uint32_t)f2bf(W[(size_t)(kb + 1) * COUT + n]) : 0u;
  Wpk[idx] = lo | (hi << 16);
}

// agg[n,c] = dense[n,c] * dinv[n]^2   (self-loop message, also zero-initializes agg)
__global__ void selfloop_init_kernel(const float* __restrict__ dense,
                                     const float* __restrict__ dinv,
                                     float* __restrict__ agg, long long total, int C) {
  long long i = (long long)blockIdx.x * blockDim.x + threadIdx.x;
  if (i >= total) return;
  int node = (int)(i / C);
  float di = dinv[node];
  agg[i] = dense[i] * di * di;
}

// agg[dst] += dense[src] * dinv[src]*dinv[dst], float2-vectorized over channels
__global__ void edge_scatter_kernel(const float* __restrict__ dense,
                                    const int* __restrict__ ei,
                                    const float* __restrict__ dinv,
                                    float* __restrict__ agg, int E, int C) {
  long long gid = (long long)blockIdx.x * blockDim.x + threadIdx.x;
  int halfC = C >> 1;
  long long total = (long long)E * halfC;
  if (gid >= total) return;
  int e  = (int)(gid / halfC);
  int c2 = (int)(gid % halfC) * 2;
  int s = ei[e];
  int d = ei[E + e];
  float w = dinv[s] * dinv[d];
  float2 h = *(const float2*)(dense + (size_t)s * C + c2);
  float* ap = agg + (size_t)d * C + c2;
  atomicAdd(ap,     h.x * w);
  atomicAdd(ap + 1, h.y * w);
}

// bias + relu, write bf16 activations with padded row stride LD (zero-fill padding cols)
__global__ void finalize_relu_bf16_kernel(const float* __restrict__ agg,
                                          const float* __restrict__ bias,
                                          unsigned short* __restrict__ out,
                                          long long total, int C, int LD) {
  long long i = (long long)blockIdx.x * blockDim.x + threadIdx.x;
  if (i >= total) return;
  int c = (int)(i % LD);
  int n = (int)(i / LD);
  if (c < C) {
    float t = agg[(size_t)n * C + c] + bias[c];
    t = t > 0.0f ? t : 0.0f;
    out[i] = f2bf(t);
  } else {
    out[i] = 0;
  }
}

__global__ void finalize_sigmoid_f32_kernel(const float* __restrict__ agg,
                                            const float* __restrict__ bias,
                                            float* __restrict__ out,
                                            long long total, int C) {
  long long i = (long long)blockIdx.x * blockDim.x + threadIdx.x;
  if (i >= total) return;
  int c = (int)(i % C);
  float t = agg[i] + bias[c];
  out[i] = 1.0f / (1.0f + expf(-t));
}

// ---------------- WMMA GEMM ----------------
// D[N,COUT](f32) = A[N,KT](bf16, K zero-padded to KT) @ Wpk (pre-packed B fragments)
// One wave: one 16-col N-tile x 5 row tiles (80 rows); N = 625 * 80 exactly.
// Depth-1 software pipeline enforced with sched_barrier: loads(kt+1) | wmma(kt).
template <int CIN, int COUT>
__global__ __launch_bounds__(128)
__attribute__((amdgpu_waves_per_eu(1)))
void gemm_wmma_bf16(const unsigned short* __restrict__ A,
                    const uint4* __restrict__ Wpk,
                    float* __restrict__ D) {
  constexpr int KT     = (CIN + 31) & ~31;
  constexpr int KSTEPS = KT / 32;
  constexpr int NT     = (COUT + 15) / 16;
  constexpr int MT     = 5;

  int lane = threadIdx.x & 31;
  int wv   = threadIdx.x >> 5;
  int nt   = blockIdx.x * 4 + wv;
  if (nt >= NT) return;
  int row0 = blockIdx.y * (MT * 16);

  int m  = lane & 15;
  int hs = (lane >> 4) << 3;                 // 0 or 8 (K-half for A, row-half for D)
  const unsigned short* arow = A + (size_t)(row0 + m) * KT + (hs << 1);
  const uint4* bbase = Wpk + ((size_t)nt * 32 + lane) * 2;

  ABu abuf[2][MT];
  ABu bbuf[2];

  auto load_frag = [&](int kt, int buf) {
    const uint4* bp = bbase + (size_t)kt * NT * 64;
    bbuf[buf].q[0] = bp[0];
    bbuf[buf].q[1] = bp[1];
#pragma unroll
    for (int t = 0; t < MT; t++) {
      const unsigned short* ap = arow + (size_t)t * 16 * KT + kt * 32;
      abuf[buf][t].q[0] = *(const uint4*)(ap);       // K = kt*32 + 2*hs + 0..7
      abuf[buf][t].q[1] = *(const uint4*)(ap + 8);   // K = kt*32 + 2*hs + 8..15
    }
  };

  v8f acc[MT];
#pragma unroll
  for (int t = 0; t < MT; t++) acc[t] = (v8f){};

  load_frag(0, 0);
#pragma unroll
  for (int kt = 0; kt < KSTEPS; kt++) {
    int cur = kt & 1;
    if (kt + 1 < KSTEPS) load_frag(kt + 1, cur ^ 1);   // prefetch next K-step
    __builtin_amdgcn_sched_barrier(0);                 // keep prefetch above the WMMAs
#pragma unroll
    for (int t = 0; t < MT; t++) {
      acc[t] = __builtin_amdgcn_wmma_f32_16x16x32_bf16(
          false, abuf[cur][t].v, false, bbuf[cur].v, (short)0, acc[t], false, false);
    }
  }

  int n = nt * 16 + (lane & 15);
  if (n < COUT) {
#pragma unroll
    for (int t = 0; t < MT; t++) {
#pragma unroll
      for (int r = 0; r < 8; r++) {
        D[(size_t)(row0 + t * 16 + hs + r) * COUT + n] = acc[t][r];
      }
    }
  }
}

// ---------------- per-layer driver ----------------
// hin: bf16 activations with row stride KT(CIN). hout written with row stride ld_out.
template <int CIN, int COUT>
static void run_layer(const unsigned short* hin, const uint4* wpk,
                      const float* bias, const int* ei, const float* dinv,
                      float* dense, float* agg,
                      unsigned short* hout_bf16, int ld_out, float* fout_f32,
                      hipStream_t stream) {
  const int N = NN, E = NE;
  constexpr int NT = (COUT + 15) / 16;
  dim3 grid((NT + 3) / 4, N / 80);            // N = 625 * 80 exactly
  gemm_wmma_bf16<CIN, COUT><<<grid, 128, 0, stream>>>(hin, wpk, dense);

  long long total = (long long)N * COUT;
  selfloop_init_kernel<<<(unsigned)((total + 255) / 256), 256, 0, stream>>>(
      dense, dinv, agg, total, COUT);

  long long etotal = (long long)E * (COUT / 2);
  edge_scatter_kernel<<<(unsigned)((etotal + 255) / 256), 256, 0, stream>>>(
      dense, ei, dinv, agg, E, COUT);

  if (hout_bf16) {
    long long otot = (long long)N * ld_out;
    finalize_relu_bf16_kernel<<<(unsigned)((otot + 255) / 256), 256, 0, stream>>>(
        agg, bias, hout_bf16, otot, COUT, ld_out);
  } else {
    finalize_sigmoid_f32_kernel<<<(unsigned)((total + 255) / 256), 256, 0, stream>>>(
        agg, bias, fout_f32, total, COUT);
  }
}

static inline int ktpad(int c) { return (c + 31) & ~31; }

// ---------------- entry point ----------------

extern "C" void kernel_launch(void* const* d_in, const int* in_sizes, int n_in,
                              void* d_out, int out_size, void* d_ws, size_t ws_size,
                              hipStream_t stream) {
  (void)in_sizes; (void)n_in; (void)out_size; (void)ws_size;
  const float* x  = (const float*)d_in[0];
  const int*   ei = (const int*)d_in[1];
  const float* We[4] = {(const float*)d_in[2], (const float*)d_in[4],
                        (const float*)d_in[6], (const float*)d_in[8]};
  const float* be[4] = {(const float*)d_in[3], (const float*)d_in[5],
                        (const float*)d_in[7], (const float*)d_in[9]};
  const float* Wn[4] = {(const float*)d_in[10], (const float*)d_in[12],
                        (const float*)d_in[14], (const float*)d_in[16]};
  const float* bn[4] = {(const float*)d_in[11], (const float*)d_in[13],
                        (const float*)d_in[15], (const float*)d_in[17]};

  const int N = NN, E = NE;
  const int CMAX = 256;
  const int edims[4][2] = {{128, 166}, {166, 192}, {192, 218}, {218, 256}};

  // workspace carve-out (~130 MB)
  char* ws = (char*)d_ws;
  size_t off = 0;
  auto carve = [&](size_t bytes) -> char* {
    char* p = ws + off;
    off = (off + bytes + 255) & ~(size_t)255;
    return p;
  };
  float*          dinv  = (float*)carve((size_t)N * 4);
  unsigned short* hbf   = (unsigned short*)carve((size_t)N * CMAX * 2);
  float*          dense = (float*)carve((size_t)N * CMAX * 4);
  float*          agg   = (float*)carve((size_t)N * CMAX * 4);

  // packed-weight buffers + pack launches
  uint4* wepk[4];
  uint4* wnpk[4];
  for (int i = 0; i < 4; i++) {
    int CIN = edims[i][0], COUT = edims[i][1];
    int ks = ktpad(CIN) / 32, nt = (COUT + 15) / 16;
    int total = ks * nt * 256;                              // dwords
    wepk[i] = (uint4*)carve((size_t)total * 4);
    pack_w_kernel<<<(total + 255) / 256, 256, 0, stream>>>(We[i], (uint32_t*)wepk[i],
                                                           CIN, COUT, nt, total);
  }
  for (int i = 0; i < 4; i++) {
    int ks = 4, nt = 8;
    int total = ks * nt * 256;
    wnpk[i] = (uint4*)carve((size_t)total * 4);
    pack_w_kernel<<<(total + 255) / 256, 256, 0, stream>>>(Wn[i], (uint32_t*)wnpk[i],
                                                           128, 128, nt, total);
  }

  // GCN norm: deg = in-degree + 1 (self loop), dinv = rsqrt(deg)
  deg_init_kernel<<<(N + 255) / 256, 256, 0, stream>>>(dinv, N);
  deg_edges_kernel<<<(E + 255) / 256, 256, 0, stream>>>(ei, dinv, E);
  dinv_finish_kernel<<<(N + 255) / 256, 256, 0, stream>>>(dinv, N);

  float* eout = (float*)d_out;                   // [N, 256]
  float* vout = (float*)d_out + (size_t)N * 256; // [N, 128]

  // edge-feature branch: 128 -> 166 -> 192 -> 218 -> 256 (relu x3, sigmoid)
  {
    long long n = (long long)N * 128;              // KT(128) == 128, no padding
    f32_to_bf16_kernel<<<(unsigned)((n + 255) / 256), 256, 0, stream>>>(x, hbf, n);
  }
  run_layer<128, 166>(hbf, wepk[0], be[0], ei, dinv, dense, agg, hbf, ktpad(166), nullptr, stream);
  run_layer<166, 192>(hbf, wepk[1], be[1], ei, dinv, dense, agg, hbf, ktpad(192), nullptr, stream);
  run_layer<192, 218>(hbf, wepk[2], be[2], ei, dinv, dense, agg, hbf, ktpad(218), nullptr, stream);
  run_layer<218, 256>(hbf, wepk[3], be[3], ei, dinv, dense, agg, nullptr, 0, eout, stream);

  // node-feature branch: 128 -> 128 x4 (relu x3, sigmoid)
  {
    long long n = (long long)N * 128;
    f32_to_bf16_kernel<<<(unsigned)((n + 255) / 256), 256, 0, stream>>>(x, hbf, n);
  }
  run_layer<128, 128>(hbf, wnpk[0], bn[0], ei, dinv, dense, agg, hbf, 128, nullptr, stream);
  run_layer<128, 128>(hbf, wnpk[1], bn[1], ei, dinv, dense, agg, hbf, 128, nullptr, stream);
  run_layer<128, 128>(hbf, wnpk[2], bn[2], ei, dinv, dense, agg, hbf, 128, nullptr, stream);
  run_layer<128, 128>(hbf, wnpk[3], bn[3], ei, dinv, dense, agg, nullptr, 0, vout, stream);
}
